// ScaledDotProduct_26216480375451
// MI455X (gfx1250) — matrix operand order repaired and checked by reading, not compiled, not run on hardware
//
#include <hip/hip_runtime.h>
#include <math.h>

typedef __attribute__((ext_vector_type(16))) _Float16 v16h;
typedef __attribute__((ext_vector_type(8)))  _Float16 v8h;
typedef __attribute__((ext_vector_type(8)))  float    v8f;

#define SEQ   2048
#define DIMH  64
#define QT    16      // q rows per wave
#define KT    32      // keys per iteration (== WMMA K of the P*V gemm)
#define NW    4       // waves per workgroup
#define SCALE 0.125f  // 1/sqrt(64)

#define SKP   72      // sK row stride (halves): 144B, 16B-aligned, bank-spread
#define SVP   48      // sVt row stride (halves): 96B, 16B-aligned
#define SPP   40      // sP row stride (halves): 80B, 16B-aligned

// ---- DPP 16-lane-row butterfly reductions (no LDS traffic) ----
template <int CTRL>
__device__ __forceinline__ float dpp_xfer(float x) {
    return __int_as_float(__builtin_amdgcn_update_dpp(
        0, __float_as_int(x), CTRL, 0xF, 0xF, true));
}
__device__ __forceinline__ float rowmax16(float x) {
    x = fmaxf(x, dpp_xfer<0xB1>(x));    // quad_perm [1,0,3,2]  : xor 1
    x = fmaxf(x, dpp_xfer<0x4E>(x));    // quad_perm [2,3,0,1]  : xor 2
    x = fmaxf(x, dpp_xfer<0x141>(x));   // row_half_mirror      : xor 7
    x = fmaxf(x, dpp_xfer<0x140>(x));   // row_mirror           : xor 15
    return x;
}
__device__ __forceinline__ float rowsum16(float x) {
    x += dpp_xfer<0xB1>(x);
    x += dpp_xfer<0x4E>(x);
    x += dpp_xfer<0x141>(x);
    x += dpp_xfer<0x140>(x);
    return x;
}

// halves p[0..7] and p[16..23] -> v16h (two aligned 16B LDS reads)
__device__ __forceinline__ v16h ld16(const _Float16* p) {
    v8h lo = *(const v8h*)p;
    v8h hi = *(const v8h*)(p + 16);
    return __builtin_shufflevector(lo, hi, 0,1,2,3,4,5,6,7,8,9,10,11,12,13,14,15);
}

__global__ __launch_bounds__(NW * 32)
void ScaledDotProduct_26216480375451_kernel(const float* __restrict__ Q,
                                            const float* __restrict__ K,
                                            const float* __restrict__ V,
                                            float* __restrict__ O)
{
    const int b    = blockIdx.y;
    const int tid  = threadIdx.x;          // 0..127
    const int w    = tid >> 5;             // wave id in block
    const int lane = tid & 31;
    const int half = lane >> 4;
    const int lid  = lane & 15;
    const int kb8  = half * 8;             // A-fragment K-base (ISA layout)
    const int kb16 = half * 16;            // unused; B fragments also use kb8
    const int q0   = (blockIdx.x * NW + w) * QT;

    const float* Qb = Q + ((size_t)b * SEQ + q0) * DIMH;
    const float* Kb = K + (size_t)b * SEQ * DIMH;
    const float* Vb = V + (size_t)b * SEQ * DIMH;
    float*       Ob = O + ((size_t)b * SEQ + q0) * DIMH;

    __shared__ _Float16 sK [KT][SKP];          // K tile, [key][d], f16
    __shared__ _Float16 sVt[DIMH][SVP];        // V tile transposed, [d][key], f16
    __shared__ _Float16 sP [NW][QT][SPP];      // per-wave P tile, [qrow][key], f16

    // ---- Q tile -> two 16x32 f16 A-fragments, pre-scaled by 1/sqrt(d) ----
    v16h qa0 = {}, qa1 = {};
    {
        const float* qr = Qb + lid * DIMH;
#pragma unroll
        for (int j = 0; j < 8; ++j) {
            qa0[j]     = (_Float16)(qr[kb8 + j]      * SCALE);
            qa0[8 + j] = (_Float16)(qr[16 + kb8 + j] * SCALE);
            qa1[j]     = (_Float16)(qr[32 + kb8 + j] * SCALE);
            qa1[8 + j] = (_Float16)(qr[48 + kb8 + j] * SCALE);
        }
    }

    v8f acc[4] = {};
    float mrow[8], lrow[8];
#pragma unroll
    for (int r = 0; r < 8; ++r) { mrow[r] = -INFINITY; lrow[r] = 0.f; }

    for (int k0 = 0; k0 < SEQ; k0 += KT) {
        // ---- cooperative staging: K tile [32][64] and V tile transposed ----
        // 2048 floats each; 128 threads * 4 float4 per array
#pragma unroll
        for (int i = 0; i < (KT * DIMH) / (NW * 32 * 4); ++i) {   // 4 iters
            const int lin = (i * NW * 32 + tid) * 4;
            const int r = lin >> 6, c = lin & 63;
            const float4 kv = *(const float4*)&Kb[(size_t)(k0 + r) * DIMH + c];
            sK[r][c + 0] = (_Float16)kv.x;
            sK[r][c + 1] = (_Float16)kv.y;
            sK[r][c + 2] = (_Float16)kv.z;
            sK[r][c + 3] = (_Float16)kv.w;
            const float4 vv = *(const float4*)&Vb[(size_t)(k0 + r) * DIMH + c];
            sVt[c + 0][r] = (_Float16)vv.x;
            sVt[c + 1][r] = (_Float16)vv.y;
            sVt[c + 2][r] = (_Float16)vv.z;
            sVt[c + 3][r] = (_Float16)vv.w;
        }
        if (k0 + KT < SEQ && tid < KT) {   // pull next tiles toward the caches
            __builtin_prefetch(&Kb[(size_t)(k0 + KT + tid) * DIMH], 0, 1);
            __builtin_prefetch(&Vb[(size_t)(k0 + KT + tid) * DIMH], 0, 1);
        }
        __syncthreads();

        // ---- S = (Q*scale) . K^T : two 16-key subtiles, D=64 as 2x K=32 ----
        v8f sS[2] = {};
#pragma unroll
        for (int s = 0; s < 2; ++s) {
            const _Float16* krow = &sK[s * 16 + lid][0];
            const v16h kf0 = ld16(krow + kb8);        // d = 0..31 chunk
            const v16h kf1 = ld16(krow + 32 + kb8);   // d = 32..63 chunk
            sS[s] = __builtin_amdgcn_wmma_f32_16x16x32_f16(false, qa0, false, kf0,
                                                           (short)0, sS[s], false, false);
            sS[s] = __builtin_amdgcn_wmma_f32_16x16x32_f16(false, qa1, false, kf1,
                                                           (short)0, sS[s], false, false);
        }

        // ---- online softmax (rows 0-7 in lanes 0-15, rows 8-15 in 16-31) ----
        float p0[8], p1[8];
#pragma unroll
        for (int r = 0; r < 8; ++r) {
            const float mx   = rowmax16(fmaxf(sS[0][r], sS[1][r]));
            const float mnew = fmaxf(mrow[r], mx);
            const float corr = __expf(mrow[r] - mnew);
            p0[r] = __expf(sS[0][r] - mnew);
            p1[r] = __expf(sS[1][r] - mnew);
            const float sum = rowsum16(p0[r] + p1[r]);
            lrow[r] = lrow[r] * corr + sum;
            mrow[r] = mnew;
            acc[0][r] *= corr; acc[1][r] *= corr;
            acc[2][r] *= corr; acc[3][r] *= corr;
        }

        // ---- stage P (16x32 f16) through LDS to re-layout as an A-fragment ----
#pragma unroll
        for (int r = 0; r < 8; ++r) {
            sP[w][kb8 + r][lid]      = (_Float16)p0[r];
            sP[w][kb8 + r][16 + lid] = (_Float16)p1[r];
        }
        // per-wave LDS: DS ops are in-order within a wave; compiler inserts waits
        v8h plo = *(const v8h*)&sP[w][lid][kb8];
        v8h phi = *(const v8h*)&sP[w][lid][16 + kb8];
        const v16h pa = __builtin_shufflevector(plo, phi,
                          0,1,2,3,4,5,6,7,8,9,10,11,12,13,14,15);

        // ---- acc += P . V : 4 WMMAs cover d = 0..63 ----
#pragma unroll
        for (int db = 0; db < 4; ++db) {
            const v16h vf = ld16(&sVt[db * 16 + lid][kb8]);
            acc[db] = __builtin_amdgcn_wmma_f32_16x16x32_f16(false, pa, false, vf,
                                                             (short)0, acc[db], false, false);
        }
        __syncthreads();   // protect sK/sVt before next iteration's staging
    }

    // ---- normalize and write out (C/D layout: M = kb8 + r, N = lid) ----
#pragma unroll
    for (int r = 0; r < 8; ++r) {
        const float inv = 1.f / lrow[r];
        const int m = kb8 + r;
#pragma unroll
        for (int db = 0; db < 4; ++db)
            Ob[(size_t)m * DIMH + db * 16 + lid] = acc[db][r] * inv;
    }
}

extern "C" void kernel_launch(void* const* d_in, const int* in_sizes, int n_in,
                              void* d_out, int out_size, void* d_ws, size_t ws_size,
                              hipStream_t stream) {
    const float* q = (const float*)d_in[0];
    const float* k = (const float*)d_in[1];
    const float* v = (const float*)d_in[2];
    float* out = (float*)d_out;
    const int batches = in_sizes[0] / (SEQ * DIMH);   // 16
    dim3 grid(SEQ / (QT * NW), batches);
    ScaledDotProduct_26216480375451_kernel<<<grid, dim3(NW * 32), 0, stream>>>(q, k, v, out);
}
